// BlockGCN_30416958390823
// MI455X (gfx1250) — matrix operand order repaired and checked by reading, not compiled, not run on hardware
//
#include <hip/hip_runtime.h>

#define NNODES 100000
#define NEDGES 1600000
#define IN_F   256
#define H_F    64
#define N_CLS  40
#define N_CLS_PAD 48

typedef __attribute__((ext_vector_type(16))) __bf16 v16bf;
typedef __attribute__((ext_vector_type(2)))  __bf16 v2bf;
typedef __attribute__((ext_vector_type(8)))  float  v8f;

// f32 -> bf16 via native fptrunc; backend picks v_cvt_*_bf16 when available.
__device__ __forceinline__ unsigned short f2bf(float f) {
    return __builtin_bit_cast(unsigned short, (__bf16)f);
}
__device__ __forceinline__ unsigned int pkbf2(float lo, float hi) {
    v2bf p = { (__bf16)lo, (__bf16)hi };
    return __builtin_bit_cast(unsigned int, p);
}

// ---------------------------------------------------------------------------
// Pack W1 [256,64] and W2 [64,40]->padded [64,48] into bf16 B-fragment order:
// entry ((kc*NT + nt)*32 + lane)*16 + i  holds  W[kc*32 + kb + (i&7) + ((i&8)?16:0)][nt*16 + (lane&15)]
// with kb = (lane&16)?8:0   (B 32x16 bf16 layout: lanes 0-15 K=0..7,16..23; 16-31 K=8..15,24..31)
// ---------------------------------------------------------------------------
__global__ void k_pack_w(const float* __restrict__ W1, const float* __restrict__ W2,
                         unsigned short* __restrict__ W1p, unsigned short* __restrict__ W2p) {
    int idx = blockIdx.x * 256 + threadIdx.x;
    const int W1N = 8 * 4 * 32 * 16;   // 16384
    const int W2N = 2 * 3 * 32 * 16;   // 3072
    if (idx < W1N) {
        int i    = idx & 15;
        int lane = (idx >> 4) & 31;
        int t    = idx >> 9;
        int nt   = t & 3;
        int kc   = t >> 2;
        int kb   = (lane & 16) ? 8 : 0;
        int K    = kc * 32 + kb + (i & 7) + ((i & 8) ? 16 : 0);
        int n    = nt * 16 + (lane & 15);
        W1p[idx] = f2bf(W1[K * H_F + n]);
    } else if (idx < W1N + W2N) {
        int j    = idx - W1N;
        int i    = j & 15;
        int lane = (j >> 4) & 31;
        int t    = j >> 9;
        int nt   = t % 3;
        int kc   = t / 3;
        int kb   = (lane & 16) ? 8 : 0;
        int K    = kc * 32 + kb + (i & 7) + ((i & 8) ? 16 : 0);
        int n    = nt * 16 + (lane & 15);
        W2p[j]   = (n < N_CLS) ? f2bf(W2[K * N_CLS + n]) : (unsigned short)0;
    }
}

__global__ void k_zero(float* __restrict__ p, int n) {
    int i = blockIdx.x * 256 + threadIdx.x;
    if (i < n) p[i] = 0.0f;
}

// ---------------------------------------------------------------------------
// support1 = x @ W1   (bf16 WMMA, f32 accum).  1 wave -> 16 rows x 64 cols.
// A 16x32 bf16 layout: lanes 0-15 hold K=kb..kb+7 (elems 0-7) and K=kb+16..kb+23
// (elems 8-15) with kb=0; lanes 16-31 same with kb=8.
// ---------------------------------------------------------------------------
__global__ void __launch_bounds__(256) k_gemm1(const float* __restrict__ x,
                                               const unsigned short* __restrict__ W1p,
                                               float* __restrict__ sup1) {
    int wave = threadIdx.x >> 5;
    int lane = threadIdx.x & 31;
    int row0 = (blockIdx.x * 8 + wave) * 16;
    if (row0 >= NNODES) return;
    int m   = lane & 15;
    int kbA = (lane & 16) ? 8 : 0;
    const float* xr = x + (size_t)(row0 + m) * IN_F;
    const v16bf* bp = (const v16bf*)W1p;

    v8f acc0 = {}, acc1 = {}, acc2 = {}, acc3 = {};
    for (int kc = 0; kc < 8; ++kc) {
        int k0 = kc * 32;
        float4 f0 = *(const float4*)(xr + k0 + kbA);
        float4 f1 = *(const float4*)(xr + k0 + kbA + 4);
        float4 f2 = *(const float4*)(xr + k0 + kbA + 16);
        float4 f3 = *(const float4*)(xr + k0 + kbA + 20);
        // build_vector of fptruncs -> packed bf16 converts where supported
        v16bf av = { (__bf16)f0.x, (__bf16)f0.y, (__bf16)f0.z, (__bf16)f0.w,
                     (__bf16)f1.x, (__bf16)f1.y, (__bf16)f1.z, (__bf16)f1.w,
                     (__bf16)f2.x, (__bf16)f2.y, (__bf16)f2.z, (__bf16)f2.w,
                     (__bf16)f3.x, (__bf16)f3.y, (__bf16)f3.z, (__bf16)f3.w };

        v16bf b0 = bp[(kc * 4 + 0) * 32 + lane];
        v16bf b1 = bp[(kc * 4 + 1) * 32 + lane];
        v16bf b2 = bp[(kc * 4 + 2) * 32 + lane];
        v16bf b3 = bp[(kc * 4 + 3) * 32 + lane];

        acc0 = __builtin_amdgcn_wmma_f32_16x16x32_bf16(false, av, false, b0, (short)0, acc0, false, false);
        acc1 = __builtin_amdgcn_wmma_f32_16x16x32_bf16(false, av, false, b1, (short)0, acc1, false, false);
        acc2 = __builtin_amdgcn_wmma_f32_16x16x32_bf16(false, av, false, b2, (short)0, acc2, false, false);
        acc3 = __builtin_amdgcn_wmma_f32_16x16x32_bf16(false, av, false, b3, (short)0, acc3, false, false);
    }
    // C/D layout: VGPR r: lanes 0-15 -> M=r, lanes 16-31 -> M=8+r; N = lane&15
    float* o = sup1 + (size_t)row0 * H_F;
    int cr = (lane & 16) ? 8 : 0;
    #pragma unroll
    for (int r = 0; r < 8; ++r) {
        size_t rowoff = (size_t)(r + cr) * H_F;
        o[rowoff +  0 + m] = acc0[r];
        o[rowoff + 16 + m] = acc1[r];
        o[rowoff + 32 + m] = acc2[r];
        o[rowoff + 48 + m] = acc3[r];
    }
}

// h_acc[dst[e]][f] += ew[e] * sup1[src[e]][f]   (4 edges x 64 feats per block)
__global__ void k_spmm1(const int* __restrict__ src, const int* __restrict__ dst,
                        const float* __restrict__ ew, const float* __restrict__ sup1,
                        float* __restrict__ hacc) {
    int e = blockIdx.x * 4 + (threadIdx.x >> 6);
    int f = threadIdx.x & 63;
    if (e >= NEDGES) return;
    float v = ew[e] * sup1[(size_t)src[e] * H_F + f];
    atomicAdd(&hacc[(size_t)dst[e] * H_F + f], v);
}

// h = relu(h_acc + b1) -> bf16, two elements per thread (packed cvt + b32 store)
__global__ void k_bias_relu(const float* __restrict__ hacc, const float* __restrict__ b1,
                            unsigned int* __restrict__ hbf2) {
    int idx = blockIdx.x * 256 + threadIdx.x;          // over N*64/2 pairs
    if (idx >= NNODES * H_F / 2) return;
    int i2 = idx * 2;
    int f  = i2 & (H_F - 1);                           // pair never crosses a row (64 even)
    float v0 = hacc[i2]     + b1[f];
    float v1 = hacc[i2 + 1] + b1[f + 1];
    v0 = v0 > 0.0f ? v0 : 0.0f;
    v1 = v1 > 0.0f ? v1 : 0.0f;
    hbf2[idx] = pkbf2(v0, v1);
}

// support2 = h @ W2 (padded to 48 cols). K=64 -> 2 steps, 3 N-tiles.
__global__ void __launch_bounds__(256) k_gemm2(const unsigned short* __restrict__ hbf,
                                               const unsigned short* __restrict__ W2p,
                                               float* __restrict__ sup2) {
    int wave = threadIdx.x >> 5;
    int lane = threadIdx.x & 31;
    int row0 = (blockIdx.x * 8 + wave) * 16;
    if (row0 >= NNODES) return;
    int m   = lane & 15;
    int kbA = (lane & 16) ? 8 : 0;
    const unsigned short* hr = hbf + (size_t)(row0 + m) * H_F;
    const v16bf* bp = (const v16bf*)W2p;

    v8f acc0 = {}, acc1 = {}, acc2 = {};
    #pragma unroll
    for (int kc = 0; kc < 2; ++kc) {
        int k0 = kc * 32;
        union { v16bf v; uint4 q[2]; } a;
        a.q[0] = *(const uint4*)(hr + k0 + kbA);
        a.q[1] = *(const uint4*)(hr + k0 + kbA + 16);

        v16bf b0 = bp[(kc * 3 + 0) * 32 + lane];
        v16bf b1 = bp[(kc * 3 + 1) * 32 + lane];
        v16bf b2 = bp[(kc * 3 + 2) * 32 + lane];

        acc0 = __builtin_amdgcn_wmma_f32_16x16x32_bf16(false, a.v, false, b0, (short)0, acc0, false, false);
        acc1 = __builtin_amdgcn_wmma_f32_16x16x32_bf16(false, a.v, false, b1, (short)0, acc1, false, false);
        acc2 = __builtin_amdgcn_wmma_f32_16x16x32_bf16(false, a.v, false, b2, (short)0, acc2, false, false);
    }
    float* o = sup2 + (size_t)row0 * N_CLS_PAD;
    int cr = (lane & 16) ? 8 : 0;
    #pragma unroll
    for (int r = 0; r < 8; ++r) {
        size_t rowoff = (size_t)(r + cr) * N_CLS_PAD;
        o[rowoff +  0 + m] = acc0[r];
        o[rowoff + 16 + m] = acc1[r];
        o[rowoff + 32 + m] = acc2[r];
    }
}

// out_acc[dst[e]][c] += ew[e] * sup2[src[e]][c], c in [0,40)
__global__ void k_spmm2(const int* __restrict__ src, const int* __restrict__ dst,
                        const float* __restrict__ ew, const float* __restrict__ sup2,
                        float* __restrict__ oacc) {
    int idx = blockIdx.x * 256 + threadIdx.x;       // E * 40 = 64,000,000
    int e = idx / N_CLS;
    int c = idx - e * N_CLS;
    if (e >= NEDGES) return;
    float v = ew[e] * sup2[(size_t)src[e] * N_CLS_PAD + c];
    atomicAdd(&oacc[(size_t)dst[e] * N_CLS + c], v);
}

// log_softmax over 40 classes, one wave32 per row (lane handles c=lane, c=32+lane for lane<8)
__global__ void __launch_bounds__(256) k_lsm(const float* __restrict__ oacc,
                                             const float* __restrict__ b2,
                                             float* __restrict__ out) {
    int wave = threadIdx.x >> 5;
    int lane = threadIdx.x & 31;
    int row = blockIdx.x * 8 + wave;
    if (row >= NNODES) return;
    const float* rp = oacc + (size_t)row * N_CLS;
    float v0 = rp[lane] + b2[lane];
    bool hi = lane < (N_CLS - 32);
    float v1 = hi ? (rp[32 + lane] + b2[32 + lane]) : -3.4e38f;
    float mx = fmaxf(v0, v1);
    #pragma unroll
    for (int o = 16; o > 0; o >>= 1) mx = fmaxf(mx, __shfl_xor(mx, o, 32));
    float s = __expf(v0 - mx) + (hi ? __expf(v1 - mx) : 0.0f);
    #pragma unroll
    for (int o = 16; o > 0; o >>= 1) s += __shfl_xor(s, o, 32);
    float lse = mx + __logf(s);
    float* op = out + (size_t)row * N_CLS;
    op[lane] = v0 - lse;
    if (hi) op[32 + lane] = v1 - lse;
}

extern "C" void kernel_launch(void* const* d_in, const int* in_sizes, int n_in,
                              void* d_out, int out_size, void* d_ws, size_t ws_size,
                              hipStream_t stream) {
    (void)in_sizes; (void)n_in; (void)out_size; (void)ws_size;
    const float* x   = (const float*)d_in[0];
    const int* src0  = (const int*)  d_in[1];
    const int* dst0  = (const int*)  d_in[2];
    const float* ew0 = (const float*)d_in[3];
    const int* src1  = (const int*)  d_in[4];
    const int* dst1  = (const int*)  d_in[5];
    const float* ew1 = (const float*)d_in[6];
    const float* W1  = (const float*)d_in[7];
    const float* b1  = (const float*)d_in[8];
    const float* W2  = (const float*)d_in[9];
    const float* b2  = (const float*)d_in[10];
    float* out = (float*)d_out;

    char* ws = (char*)d_ws;
    float*          sup1 = (float*)(ws + 0);                 // 25,600,000 B ; later reused as sup2
    float*          hacc = (float*)(ws + 25600000);          // 25,600,000 B ; later reused as out_acc
    unsigned short* hbf  = (unsigned short*)(ws + 51200000); // 12,800,000 B
    unsigned short* W1p  = (unsigned short*)(ws + 64000000); // 32,768 B
    unsigned short* W2p  = (unsigned short*)(ws + 64032768); //  6,144 B
    float*          sup2 = sup1;   // alias: sup1 dead after k_spmm1
    float*          oacc = hacc;   // alias: hacc dead after k_bias_relu

    k_pack_w   <<<76, 256, 0, stream>>>(W1, W2, W1p, W2p);
    k_zero     <<<(NNODES * H_F + 255) / 256, 256, 0, stream>>>(hacc, NNODES * H_F);
    k_gemm1    <<<(NNODES / 16 + 7) / 8, 256, 0, stream>>>(x, W1p, sup1);
    k_spmm1    <<<NEDGES / 4, 256, 0, stream>>>(src0, dst0, ew0, sup1, hacc);
    k_bias_relu<<<(NNODES * H_F / 2 + 255) / 256, 256, 0, stream>>>(hacc, b1, (unsigned int*)hbf);
    k_zero     <<<(NNODES * N_CLS + 255) / 256, 256, 0, stream>>>(oacc, NNODES * N_CLS);
    k_gemm2    <<<(NNODES / 16 + 7) / 8, 256, 0, stream>>>(hbf, W2p, sup2);
    k_spmm2    <<<(NEDGES * (long long)N_CLS + 255) / 256, 256, 0, stream>>>(src1, dst1, ew1, sup2, oacc);
    k_lsm      <<<(NNODES + 7) / 8, 256, 0, stream>>>(oacc, b2, out);
}
